// IntelXpuQuantLinearFp8_48473000903246
// MI455X (gfx1250) — matrix operand order repaired and checked by reading, not compiled, not run on hardware
//
#include <hip/hip_runtime.h>

// ---------------------------------------------------------------------------
// out[M,N] = x[M,K] (f16) @ dequant(W[N,K] f32 -> f16 * scale f16)^T + bias
// M=2048, K=4096, N=10240.
// v4: operand-swapped WMMA (A<-W, B<-X) so D gives 8 consecutive n per lane ->
//     vectorized b128 epilogue stores.  64x64 wave tile, WG 128x256, BK=32
//     double-buffered LDS, 1 barrier/slab, register-pipelined global loads,
//     packed dequant.  CDNA5 / gfx1250, wave32.
// ---------------------------------------------------------------------------

typedef _Float16 v4h  __attribute__((ext_vector_type(4)));
typedef _Float16 v8h  __attribute__((ext_vector_type(8)));
typedef _Float16 v16h __attribute__((ext_vector_type(16)));
typedef float    v4f  __attribute__((ext_vector_type(4)));
typedef float    v8f  __attribute__((ext_vector_type(8)));

constexpr int GEMM_M = 4 * 512;   // 2048
constexpr int GEMM_N = 10240;
constexpr int GEMM_K = 4096;

constexpr int BM = 128;            // workgroup tile M (x rows)
constexpr int BN = 256;            // workgroup tile N (output channels)
constexpr int BK = 32;             // K slab (1 WMMA k-step), double-buffered
constexpr int LDSS  = BK + 8;      // 40 halves = 80B stride -> conflict-free
constexpr int A_BUF = BM * LDSS;   // halves per X buffer
constexpr int B_BUF = BN * LDSS;   // halves per W buffer

__global__ __launch_bounds__(256)
void qlinear_fp8_wmma(const _Float16* __restrict__ X,      // [M,K] fp16
                      const float*    __restrict__ W,      // [N,K] fp32
                      const float*    __restrict__ Wscale, // [N]   fp32
                      const _Float16* __restrict__ Bias,   // [N]   fp16
                      _Float16*       __restrict__ Out)    // [M,N] fp16
{
    __shared__ _Float16 Xs[2 * A_BUF];
    __shared__ _Float16 Ws[2 * B_BUF];

    const int tid  = threadIdx.x;
    const int lane = tid & 31;          // wave32
    const int wave = tid >> 5;          // 0..7
    const int lr   = lane & 15;
    const int lh   = lane >> 4;

    const int wgM = blockIdx.y * BM;
    const int wgN = blockIdx.x * BN;

    // 2 (M) x 4 (N) wave grid; each wave: 64(m) x 64(n) = 4x4 WMMA accs
    const int waveM = (wave & 1) * 64;
    const int waveN = (wave >> 1) * 64;

    // ---- fill mapping: thread pair owns a row, 16 elements each; W covers rows r, r+128 ----
    const int fRow = tid >> 1;           // 0..127
    const int fCol = (tid & 1) * 16;     // element offset within 32-wide slab

    const _Float16 hs1 = (_Float16)Wscale[wgN + fRow];
    const _Float16 hs2 = (_Float16)Wscale[wgN + fRow + 128];
    const v4h hs1v = {hs1, hs1, hs1, hs1};
    const v4h hs2v = {hs2, hs2, hs2, hs2};

    const _Float16* aSrc  = X + (size_t)(wgM + fRow) * GEMM_K + fCol;
    const float*    bSrc1 = W + (size_t)(wgN + fRow) * GEMM_K + fCol;
    const float*    bSrc2 = W + (size_t)(wgN + fRow + 128) * GEMM_K + fCol;
    _Float16* aDst  = &Xs[fRow * LDSS + fCol];
    _Float16* bDst1 = &Ws[fRow * LDSS + fCol];
    _Float16* bDst2 = &Ws[(fRow + 128) * LDSS + fCol];

    // acc[i][j]: i = n-block (A operand = W), j = m-block (B operand = X)
    v8f acc[4][4];
#pragma unroll
    for (int i = 0; i < 4; ++i)
#pragma unroll
        for (int j = 0; j < 4; ++j)
#pragma unroll
            for (int v = 0; v < 8; ++v)
                acc[i][j][v] = 0.0f;

    // ---- prologue: slab 0 -> regs -> LDS buf0 ----
    v8h aReg[2];
    v4f bReg[2][4];
#pragma unroll
    for (int i = 0; i < 2; ++i) aReg[i] = *(const v8h*)(aSrc + i * 8);
#pragma unroll
    for (int i = 0; i < 4; ++i) {
        bReg[0][i] = *(const v4f*)(bSrc1 + i * 4);
        bReg[1][i] = *(const v4f*)(bSrc2 + i * 4);
    }
#pragma unroll
    for (int i = 0; i < 2; ++i) *(v8h*)(aDst + i * 8) = aReg[i];
#pragma unroll
    for (int i = 0; i < 4; ++i) {
        *(v4h*)(bDst1 + i * 4) = __builtin_convertvector(bReg[0][i], v4h) * hs1v;
        *(v4h*)(bDst2 + i * 4) = __builtin_convertvector(bReg[1][i], v4h) * hs2v;
    }
    __syncthreads();

    for (int k0 = 0; k0 < GEMM_K; k0 += BK) {
        const int cur  = (k0 >> 5) & 1;
        const int nxt  = cur ^ 1;
        const int kn   = k0 + BK;
        const bool more = kn < GEMM_K;

        // ---- issue next-slab global loads (in flight across compute) ----
        if (more) {
#pragma unroll
            for (int i = 0; i < 2; ++i)
                aReg[i] = *(const v8h*)(aSrc + kn + i * 8);
#pragma unroll
            for (int i = 0; i < 4; ++i) {
                bReg[0][i] = *(const v4f*)(bSrc1 + kn + i * 4);
                bReg[1][i] = *(const v4f*)(bSrc2 + kn + i * 4);
            }
            __builtin_prefetch(bSrc1 + kn + BK, 0, 3);
            __builtin_prefetch(bSrc2 + kn + BK, 0, 3);
            __builtin_prefetch(aSrc + kn + BK, 0, 3);
        }

        // ---- compute slab from buf[cur]: 16 WMMAs ----
        {
            const _Float16* Xb = &Xs[cur * A_BUF];
            const _Float16* Wb = &Ws[cur * B_BUF];

            // A operand <- W tile: A 16x32 layout: lane lr = weight row;
            // halves 0..7 at K=lh*8, halves 8..15 at K=16+lh*8
            v16h wfrag[4];
#pragma unroll
            for (int i = 0; i < 4; ++i) {
                const _Float16* p = Wb + (waveN + 16 * i + lr) * LDSS + lh * 8;
                v8h lo = *(const v8h*)p;
                v8h hi = *(const v8h*)(p + 16);
                wfrag[i] = __builtin_shufflevector(lo, hi,
                    0,1,2,3,4,5,6,7,8,9,10,11,12,13,14,15);
            }
            // B operand <- X tile: B 32x16 layout: lane lr = x row (column);
            // halves 0..7 at K=lh*8, halves 8..15 at K=16+lh*8
#pragma unroll
            for (int j = 0; j < 4; ++j) {
                const _Float16* p = Xb + (waveM + 16 * j + lr) * LDSS + lh * 8;
                v8h lo = *(const v8h*)p;
                v8h hi = *(const v8h*)(p + 16);
                v16h xfrag = __builtin_shufflevector(lo, hi,
                    0,1,2,3,4,5,6,7,8,9,10,11,12,13,14,15);
#pragma unroll
                for (int i = 0; i < 4; ++i)
                    acc[i][j] = __builtin_amdgcn_wmma_f32_16x16x32_f16(
                        /*neg_a=*/false, wfrag[i],
                        /*neg_b=*/false, xfrag,
                        /*c_mod=*/(short)0, acc[i][j],
                        /*reuse_a=*/false, /*reuse_b=*/false);
            }
        }

        // ---- regs -> LDS buf[nxt] (other buffer: safe before the barrier) ----
        if (more) {
            _Float16* aD  = aDst  + nxt * A_BUF;
            _Float16* bD1 = bDst1 + nxt * B_BUF;
            _Float16* bD2 = bDst2 + nxt * B_BUF;
#pragma unroll
            for (int i = 0; i < 2; ++i) *(v8h*)(aD + i * 8) = aReg[i];
#pragma unroll
            for (int i = 0; i < 4; ++i) {
                *(v4h*)(bD1 + i * 4) =
                    __builtin_convertvector(bReg[0][i], v4h) * hs1v;
                *(v4h*)(bD2 + i * 4) =
                    __builtin_convertvector(bReg[1][i], v4h) * hs2v;
            }
        }
        __syncthreads();
    }

    // ---- epilogue: D layout (A=W,B=X): lane lr = m row, acc VGPR v ->
    //      n = 16*i + 8*lh + v  => 8 consecutive n per lane: b128 stores ----
    v8f biasf[4];
#pragma unroll
    for (int i = 0; i < 4; ++i) {
        const int nb = wgN + waveN + 16 * i + 8 * lh;
        v8h bh = *(const v8h*)(Bias + nb);
        biasf[i] = __builtin_convertvector(bh, v8f);
    }
#pragma unroll
    for (int j = 0; j < 4; ++j) {
        const int m = wgM + waveM + 16 * j + lr;
#pragma unroll
        for (int i = 0; i < 4; ++i) {
            const int nb = wgN + waveN + 16 * i + 8 * lh;
            v8f r = acc[i][j] + biasf[i];
            v8h o = __builtin_convertvector(r, v8h);
            __builtin_nontemporal_store(
                o, (v8h*)(Out + (size_t)m * GEMM_N + nb));
        }
    }
}

extern "C" void kernel_launch(void* const* d_in, const int* in_sizes, int n_in,
                              void* d_out, int out_size, void* d_ws, size_t ws_size,
                              hipStream_t stream) {
    const _Float16* X   = (const _Float16*)d_in[0];   // [B,S,DIN] fp16
    const float*    W   = (const float*)d_in[1];      // [DOUT,DIN] fp32
    const float*    Ws  = (const float*)d_in[2];      // [DOUT,1] fp32
    const _Float16* Bv  = (const _Float16*)d_in[3];   // [DOUT] fp16
    _Float16*       Out = (_Float16*)d_out;           // [B,S,DOUT] fp16

    dim3 grid(GEMM_N / BN, GEMM_M / BM);  // 40 x 16
    dim3 block(256);
    qlinear_fp8_wmma<<<grid, block, 0, stream>>>(X, W, Ws, Bv, Out);
}
